// SparseConvNet_64_5781025980666
// MI455X (gfx1250) — matrix operand order repaired and checked by reading, not compiled, not run on hardware
//
#include <hip/hip_runtime.h>
#include <hip/hip_bf16.h>

// MI455X (gfx1250) implementation of the 17-layer masked 3D conv trunk.
// bf16 activations/weights channel-last, f32 accumulation via
// v_wmma_f32_16x16x32_bf16 (M=cout, N=16 voxels, K=2 taps x 16 cin).
// Working set (~60MB bf16) is L2-resident (192MB) -> compute-bound at 96^3.
// Loop body is fully branchless: per-tile 27-bit tap-inbounds bitmask,
// 32-bit saddr-offset selects to a zero voxel, LDS-staged weights,
// 2 voxel-tiles per wave sharing A-fragments (2 independent WMMA chains),
// and an explicit 1-deep software pipeline (fetch tp+1 before WMMA tp).

typedef __attribute__((ext_vector_type(16))) __bf16 v16bf;
typedef __attribute__((ext_vector_type(8)))  __bf16 v8bf;
typedef __attribute__((ext_vector_type(8)))  float  v8f;
typedef __attribute__((ext_vector_type(4)))  float  v4f;

#define NCONV 17
#define NTP   14   // tap pairs: ceil(27/2)

// ---------------------------------------------------------------------------
// Pack W[L][cout][cin][3][3][3] (f32) into per-lane WMMA A-fragments (bf16).
// pw[((L*NTP+tp)*32 + lane)*16 + j]:
//   h=lane>>4, cout=lane&15
//   tap=2tp+h, cin=j               (K = 16h+j)
__global__ void pack_weights_kernel(const float* __restrict__ W,
                                    __bf16* __restrict__ pw) {
  int idx = blockIdx.x * blockDim.x + threadIdx.x;
  const int total = NCONV * NTP * 32 * 16;
  if (idx >= total) return;
  int j    = idx & 15;
  int lane = (idx >> 4) & 31;
  int tp   = (idx >> 9) % NTP;
  int L    = (idx >> 9) / NTP;
  int h    = lane >> 4;
  int cout = lane & 15;
  int tap  = 2 * tp + h;
  int cin  = j;
  float v = 0.0f;
  if (tap < 27) v = W[((L * 16 + cout) * 16 + cin) * 27 + tap];
  pw[idx] = (__bf16)v;
}

__global__ void make_mask_kernel(const int* __restrict__ mi,
                                 float* __restrict__ m, int vol) {
  int v = blockIdx.x * blockDim.x + threadIdx.x;
  if (v < vol) m[v] = (mi[v] == 0) ? 1.0f : 0.0f;
}

__global__ void x_to_cl_kernel(const float* __restrict__ x,
                               __bf16* __restrict__ act, int vol) {
  int idx = blockIdx.x * blockDim.x + threadIdx.x;
  if (idx >= vol * 16) return;
  int c = idx & 15;
  int v = idx >> 4;
  act[idx] = (__bf16)x[c * vol + v];
}

__global__ void zero_bf16_kernel(__bf16* __restrict__ p, int n) {
  int i = blockIdx.x * blockDim.x + threadIdx.x;
  if (i < n) p[i] = (__bf16)0.0f;
}

// max-pool mask: window 3, stride 2, pad 1 (init 0, like lax.reduce_window)
__global__ void mask_down_kernel(const float* __restrict__ in,
                                 float* __restrict__ out,
                                 int Din, int Dout) {
  int v = blockIdx.x * blockDim.x + threadIdx.x;
  int volOut = Dout * Dout * Dout;
  if (v >= volOut) return;
  int z = v / (Dout * Dout);
  int r = v - z * Dout * Dout;
  int y = r / Dout;
  int x = r - y * Dout;
  float mx = 0.0f;
  for (int dz = 0; dz < 3; ++dz)
    for (int dy = 0; dy < 3; ++dy)
      for (int dx = 0; dx < 3; ++dx) {
        int sz = 2 * z + dz - 1, sy = 2 * y + dy - 1, sx = 2 * x + dx - 1;
        if ((unsigned)sz < (unsigned)Din && (unsigned)sy < (unsigned)Din &&
            (unsigned)sx < (unsigned)Din)
          mx = fmaxf(mx, in[(sz * Din + sy) * Din + sx]);
      }
  out[v] = mx;
}

// ---------------------------------------------------------------------------
struct TileCtx {
  int zo[3], yo[3], xo[3];  // sz*Din^2, sy*Din, sx (voxel units)
  unsigned mh;              // 27-bit tap-inbounds mask, pre-shifted by h
  bool valid;
  int vv;                   // clamped output voxel
};

__device__ __forceinline__ TileCtx make_ctx(int tile, int n, int h, int Dout,
                                            int volOut, int stride, int Din) {
  TileCtx c;
  int vox = tile * 16 + n;
  c.valid = (vox < volOut);
  c.vv = c.valid ? vox : 0;
  int oz = c.vv / (Dout * Dout);
  int rem = c.vv - oz * Dout * Dout;
  int oy = rem / Dout;
  int ox = rem - oy * Dout;
  int bz = stride * oz - 1, by = stride * oy - 1, bx = stride * ox - 1;
  int Din2 = Din * Din;
  unsigned xm = 0, ym = 0, zm = 0;
#pragma unroll
  for (int d = 0; d < 3; ++d) {
    int sz = bz + d, sy = by + d, sx = bx + d;
    c.zo[d] = sz * Din2;
    c.yo[d] = sy * Din;
    c.xo[d] = sx;
    if ((unsigned)sx < (unsigned)Din) xm |= 1u << d;
    if ((unsigned)sy < (unsigned)Din) ym |= 1u << (3 * d);
    if ((unsigned)sz < (unsigned)Din) zm |= 1u << (9 * d);
  }
  // tap t = 9dz+3dy+dx in-bounds mask; binary multiplies have no carries
  // (3-bit x-groups inside 9-bit y-planes inside 27-bit volume).
  unsigned m27 = (xm * ym) * zm;
  c.mh = c.valid ? (m27 >> h) : 0u;
  return c;
}

// Branchless B-fragment fetch. The lane's tap is T0+h; in-bounds is a
// compile-time-constant bit test on the pre-shifted mask. OOB lanes get a
// 32-bit relative offset to a zero voxel in the same workspace, keeping
// global loads in saddr+voffset form.
template <int T0, int T1>
__device__ __forceinline__ v16bf load_bfrag(const TileCtx& c, int h,
                                            const __bf16* __restrict__ actIn,
                                            unsigned zvRel) {
  constexpr int U1 = (T1 < 27) ? T1 : T0;   // tap 27 never in-bounds via mask
  constexpr int z0 = T0 / 9, y0 = (T0 / 3) % 3, x0 = T0 % 3;
  constexpr int z1 = U1 / 9, y1 = (U1 / 3) % 3, x1 = U1 % 3;
  int zs = (z0 == z1) ? c.zo[z0] : (h ? c.zo[z1] : c.zo[z0]);
  int ys = (y0 == y1) ? c.yo[y0] : (h ? c.yo[y1] : c.yo[y0]);
  int xs = (x0 == x1) ? c.xo[x0] : (h ? c.xo[x1] : c.xo[x0]);
  unsigned boff = (unsigned)(zs + ys + xs) << 5;   // voxel -> byte offset
  bool inb = (c.mh & (1u << T0)) != 0;
  boff = inb ? boff : zvRel;
  return *(const v16bf*)((const char*)actIn + boff);
}

// Fetch one pipeline stage: LDS weight fragment + both tiles' act fragments.
__device__ __forceinline__ void fetch_stage(int tp, const __bf16* wlds, int lane,
                                            const TileCtx& c0, const TileCtx& c1,
                                            int h, const __bf16* __restrict__ actIn,
                                            unsigned zvRel,
                                            v16bf& wf, v16bf& af0, v16bf& af1) {
  wf = *(const v16bf*)(wlds + ((tp * 32 + lane) << 4));
  switch (tp) {
#define CASE(TP) case TP: \
    af0 = load_bfrag<2*TP, 2*TP+1>(c0, h, actIn, zvRel); \
    af1 = load_bfrag<2*TP, 2*TP+1>(c1, h, actIn, zvRel); break;
    CASE(0) CASE(1) CASE(2) CASE(3) CASE(4) CASE(5) CASE(6)
    CASE(7) CASE(8) CASE(9) CASE(10) CASE(11) CASE(12) CASE(13)
#undef CASE
    default: af0 = v16bf{}; af1 = v16bf{}; break;
  }
}

__device__ __forceinline__ void epilogue(const v8f& acc, const TileCtx& c,
                                         int c0, float mval,
                                         const v4f& sLo, const v4f& sHi,
                                         const v4f& bLo, const v4f& bHi,
                                         __bf16* __restrict__ actOut,
                                         float* __restrict__ outF32,
                                         int volOut) {
  float o[8];
#pragma unroll
  for (int r = 0; r < 8; ++r) {
    float s = (r < 4) ? sLo[r & 3] : sHi[r & 3];
    float b = (r < 4) ? bLo[r & 3] : bHi[r & 3];
    o[r] = fmaxf(fmaf(acc[r], s, b), 0.0f) * mval;
  }
  if (c.valid) {
    v8bf ob;
#pragma unroll
    for (int r = 0; r < 8; ++r) ob[r] = (__bf16)o[r];
    *(v8bf*)(actOut + (c.vv << 4) + c0) = ob;
    if (outF32) {
#pragma unroll
      for (int r = 0; r < 8; ++r) outF32[(c0 + r) * volOut + c.vv] = o[r];
    }
  }
}

// One conv layer: each wave produces TWO 16-voxel tiles (shared A-fragments,
// two independent WMMA accumulator chains). Weights staged in LDS (14KB).
__global__ void __launch_bounds__(256)
conv3x3_wmma_kernel(const __bf16* __restrict__ actIn,
                    __bf16* __restrict__ actOut,
                    const __bf16* __restrict__ pw,
                    const float* __restrict__ bnS,
                    const float* __restrict__ bnB,
                    const float* __restrict__ mask,
                    unsigned zvRel,                 // byte offset actIn->zeroVox
                    int Din, int Dout, int stride,
                    float* __restrict__ outF32) {
  __shared__ __bf16 wlds[NTP * 32 * 16];
  {
    const uint4* src = (const uint4*)pw;
    uint4* dst = (uint4*)wlds;
#pragma unroll 2
    for (int i = threadIdx.x; i < NTP * 64; i += 256) dst[i] = src[i];
  }
  __syncthreads();

  const int lane = threadIdx.x & 31;
  const int wave = threadIdx.x >> 5;
  const int h    = lane >> 4;
  const int n    = lane & 15;
  const int volOut = Dout * Dout * Dout;

  const int tile0 = (blockIdx.x * 8 + wave) * 2;
  TileCtx c0 = make_ctx(tile0 + 0, n, h, Dout, volOut, stride, Din);
  TileCtx c1 = make_ctx(tile0 + 1, n, h, Dout, volOut, stride, Din);

  v8f acc0 = {};
  v8f acc1 = {};
  // software pipeline: fragments for tp are fetched during WMMAs of tp-1
  v16bf wf, af0, af1;
  fetch_stage(0, wlds, lane, c0, c1, h, actIn, zvRel, wf, af0, af1);
#pragma unroll
  for (int tp = 0; tp < NTP; ++tp) {
    v16bf wfn = {}, af0n = {}, af1n = {};
    if (tp + 1 < NTP)
      fetch_stage(tp + 1, wlds, lane, c0, c1, h, actIn, zvRel, wfn, af0n, af1n);
    acc0 = __builtin_amdgcn_wmma_f32_16x16x32_bf16(
        false, wf, false, af0, (short)0, acc0, false, false);
    acc1 = __builtin_amdgcn_wmma_f32_16x16x32_bf16(
        false, wf, false, af1, (short)0, acc1, false, false);
    wf = wfn; af0 = af0n; af1 = af1n;
  }

  const int cc = 8 * h;
  v4f sLo = *(const v4f*)(bnS + cc);
  v4f sHi = *(const v4f*)(bnS + cc + 4);
  v4f bLo = *(const v4f*)(bnB + cc);
  v4f bHi = *(const v4f*)(bnB + cc + 4);
  float m0 = c0.valid ? mask[c0.vv] : 0.0f;
  float m1 = c1.valid ? mask[c1.vv] : 0.0f;
  epilogue(acc0, c0, cc, m0, sLo, sHi, bLo, bHi, actOut, outF32, volOut);
  epilogue(acc1, c1, cc, m1, sLo, sHi, bLo, bHi, actOut, outF32, volOut);
}

// ---------------------------------------------------------------------------
extern "C" void kernel_launch(void* const* d_in, const int* in_sizes, int n_in,
                              void* d_out, int out_size, void* d_ws, size_t ws_size,
                              hipStream_t stream) {
  const float* x        = (const float*)d_in[0];
  const int*   mask_idx = (const int*)d_in[1];
  const float* W        = (const float*)d_in[2];
  const float* bnS      = (const float*)d_in[3];
  const float* bnB      = (const float*)d_in[4];
  float* out = (float*)d_out;

  const size_t vol96 = 96ull * 96 * 96;
  const size_t vol48 = 48ull * 48 * 48;
  const size_t vol24 = 24ull * 24 * 24;
  const size_t vol12 = 12ull * 12 * 12;
  const size_t vol6  = 6ull * 6 * 6;

  char* ws = (char*)d_ws;
  size_t off = 0;
  auto take = [&](size_t bytes) -> char* {
    char* p = ws + off;
    off += (bytes + 255) & ~(size_t)255;
    return p;
  };
  __bf16* actA = (__bf16*)take(vol96 * 16 * sizeof(__bf16));
  __bf16* actB = (__bf16*)take(vol96 * 16 * sizeof(__bf16));
  __bf16* pw   = (__bf16*)take((size_t)NCONV * NTP * 32 * 16 * sizeof(__bf16));
  __bf16* zv   = (__bf16*)take(32 * sizeof(__bf16));   // zero voxel
  float* m96 = (float*)take(vol96 * sizeof(float));
  float* m48 = (float*)take(vol48 * sizeof(float));
  float* m24 = (float*)take(vol24 * sizeof(float));
  float* m12 = (float*)take(vol12 * sizeof(float));
  float* m6  = (float*)take(vol6 * sizeof(float));

  // --- preparation ---
  {
    int total = NCONV * NTP * 32 * 16;
    pack_weights_kernel<<<(total + 255) / 256, 256, 0, stream>>>(W, pw);
  }
  zero_bf16_kernel<<<1, 32, 0, stream>>>(zv, 32);
  make_mask_kernel<<<(int)((vol96 + 255) / 256), 256, 0, stream>>>(mask_idx, m96, (int)vol96);
  x_to_cl_kernel<<<(int)((vol96 * 16 + 255) / 256), 256, 0, stream>>>(x, actA, (int)vol96);
  mask_down_kernel<<<(int)((vol48 + 255) / 256), 256, 0, stream>>>(m96, m48, 96, 48);
  mask_down_kernel<<<(int)((vol24 + 255) / 256), 256, 0, stream>>>(m48, m24, 48, 24);
  mask_down_kernel<<<(int)((vol12 + 255) / 256), 256, 0, stream>>>(m24, m12, 24, 12);
  mask_down_kernel<<<(int)((vol6 + 255) / 256), 256, 0, stream>>>(m12, m6, 12, 6);

  auto conv = [&](const __bf16* in, __bf16* o, int L, const float* mk,
                  int Din, int Dout, int stride, float* of) {
    int vol = Dout * Dout * Dout;
    int tiles = (vol + 15) / 16;
    int blocks = (tiles + 15) / 16;  // 8 waves x 2 tiles per 256-thread block
    unsigned zrel = (unsigned)((const char*)zv - (const char*)in);
    conv3x3_wmma_kernel<<<blocks, 256, 0, stream>>>(
        in, o, pw + (size_t)L * NTP * 32 * 16, bnS + L * 16, bnB + L * 16,
        mk, zrel, Din, Dout, stride, of);
  };

  float* out1 = out;
  float* out2 = out1 + 16 * vol48;
  float* out3 = out2 + 16 * vol24;
  float* out4 = out3 + 16 * vol12;

  // --- network (ping-pong actA/actB, channel-last bf16) ---
  conv(actA, actB, 0, m96, 96, 96, 1, nullptr);
  conv(actB, actA, 1, m96, 96, 96, 1, nullptr);
  conv(actA, actB, 2, m48, 96, 48, 2, nullptr);   // down -> 48^3
  conv(actB, actA, 3, m48, 48, 48, 1, nullptr);
  conv(actA, actB, 4, m48, 48, 48, 1, out1);      // net1
  conv(actB, actA, 5, m24, 48, 24, 2, nullptr);   // down -> 24^3
  conv(actA, actB, 6, m24, 24, 24, 1, nullptr);
  conv(actB, actA, 7, m24, 24, 24, 1, nullptr);
  conv(actA, actB, 8, m24, 24, 24, 1, out2);      // net2
  conv(actB, actA, 9, m12, 24, 12, 2, nullptr);   // down -> 12^3
  conv(actA, actB, 10, m12, 12, 12, 1, nullptr);
  conv(actB, actA, 11, m12, 12, 12, 1, nullptr);
  conv(actA, actB, 12, m12, 12, 12, 1, out3);     // net3
  conv(actB, actA, 13, m6, 12, 6, 2, nullptr);    // down -> 6^3
  conv(actA, actB, 14, m6, 6, 6, 1, nullptr);
  conv(actB, actA, 15, m6, 6, 6, 1, nullptr);
  conv(actA, actB, 16, m6, 6, 6, 1, out4);        // net4
}
